// BINLayer_77232101917001
// MI455X (gfx1250) — 8-bit matrix operand order repaired and checked by reading, compiled, not run on hardware
//
#include <hip/hip_runtime.h>


#define DDIM  4096
#define BROWS 8192

typedef int   v16i __attribute__((ext_vector_type(16)));
typedef float v8f  __attribute__((ext_vector_type(8)));

#define FP8_P1 0x38u
#define FP8_M1 0xB8u

__device__ __forceinline__ unsigned pack4sign(float4 f) {
    unsigned b0 = (f.x >= 0.0f) ? FP8_P1 : FP8_M1;
    unsigned b1 = (f.y >= 0.0f) ? FP8_P1 : FP8_M1;
    unsigned b2 = (f.z >= 0.0f) ? FP8_P1 : FP8_M1;
    unsigned b3 = (f.w >= 0.0f) ? FP8_P1 : FP8_M1;
    return b0 | (b1 << 8) | (b2 << 16) | (b3 << 24);
}

__global__ __launch_bounds__(256) void bin_x_kernel(const float* __restrict__ x,
                                                    unsigned char* __restrict__ x8) {
    size_t i = ((size_t)blockIdx.x * blockDim.x + threadIdx.x) * 16;
    const float4* xf = (const float4*)(x + i);
    uint4 o;
    o.x = pack4sign(xf[0]);
    o.y = pack4sign(xf[1]);
    o.z = pack4sign(xf[2]);
    o.w = pack4sign(xf[3]);
    *(uint4*)(x8 + i) = o;
}

__global__ __launch_bounds__(256) void bin_wT_kernel(const float* __restrict__ W,
                                                     unsigned char* __restrict__ w8t) {
    __shared__ unsigned char tile[32][33];
    const int kBase = blockIdx.x * 32;
    const int nBase = blockIdx.y * 32;
#pragma unroll
    for (int i = 0; i < 4; ++i) {
        int r = threadIdx.y * 4 + i;
        float v = W[(size_t)(kBase + r) * DDIM + nBase + threadIdx.x];
        tile[r][threadIdx.x] = (v >= 0.0f) ? (unsigned char)FP8_P1 : (unsigned char)FP8_M1;
    }
    __syncthreads();
#pragma unroll
    for (int i = 0; i < 4; ++i) {
        int r = threadIdx.y * 4 + i;
        w8t[(size_t)(nBase + r) * DDIM + kBase + threadIdx.x] = tile[threadIdx.x][r];
    }
}

__global__ __launch_bounds__(256) void bin_gemm_kernel(const unsigned char* __restrict__ X8,
                                                       const unsigned char* __restrict__ W8t,
                                                       const float* __restrict__ bias,
                                                       float* __restrict__ out) {
    const int lane = threadIdx.x & 31;
    const int wave = threadIdx.x >> 5;
    const int wm   = wave >> 1;
    const int wn   = wave & 1;
    const int mBase = blockIdx.x * 128 + wm * 32;
    const int nBase = blockIdx.y * 128 + wn * 64;
    const int half = lane >> 4;
    const int l16  = lane & 15;

    const unsigned char* aRow[2];
    aRow[0] = X8 + (size_t)(mBase + l16)      * DDIM;
    aRow[1] = X8 + (size_t)(mBase + 16 + l16) * DDIM;
    const unsigned char* bCol[4];
#pragma unroll
    for (int nt = 0; nt < 4; ++nt)
        bCol[nt] = W8t + (size_t)(nBase + nt * 16 + l16) * DDIM;

    v8f acc[2][4] = {};

    for (int k0 = 0; k0 < DDIM; k0 += 128) {
#pragma unroll
        for (int mt = 0; mt < 2; ++mt) __builtin_prefetch(aRow[mt] + k0 + 128, 0, 1);
#pragma unroll
        for (int nt = 0; nt < 4; ++nt) __builtin_prefetch(bCol[nt] + k0 + 128, 0, 1);

        v16i a[2], b[4];
#pragma unroll
        for (int c = 0; c < 8; ++c) {
            const int off = k0 + ((c >> 1) << 5) + (half << 4) + ((c & 1) << 3);
#pragma unroll
            for (int mt = 0; mt < 2; ++mt) {
                uint2 t = *(const uint2*)(aRow[mt] + off);
                a[mt][2 * c] = (int)t.x;
                a[mt][2 * c + 1] = (int)t.y;
            }
        }
#pragma unroll
        for (int g = 0; g < 4; ++g) {
            const int off = k0 + (g << 5) + (half << 4);
#pragma unroll
            for (int nt = 0; nt < 4; ++nt) {
                uint4 t = *(const uint4*)(bCol[nt] + off);
                b[nt][4 * g]     = (int)t.x;
                b[nt][4 * g + 1] = (int)t.y;
                b[nt][4 * g + 2] = (int)t.z;
                b[nt][4 * g + 3] = (int)t.w;
            }
        }

#pragma unroll
        for (int mt = 0; mt < 2; ++mt)
#pragma unroll
            for (int nt = 0; nt < 4; ++nt)
                acc[mt][nt] = __builtin_amdgcn_wmma_f32_16x16x128_fp8_fp8(
                    a[mt], b[nt], (short)0, acc[mt][nt], false, false);
    }

    float bs[4];
#pragma unroll
    for (int nt = 0; nt < 4; ++nt)
        bs[nt] = (bias[nBase + nt * 16 + l16] >= 0.0f) ? 1.0f : -1.0f;

#pragma unroll
    for (int mt = 0; mt < 2; ++mt) {
#pragma unroll
        for (int r = 0; r < 8; ++r) {
            const int m = mBase + mt * 16 + half * 8 + r;
            float* o = out + (size_t)m * DDIM + nBase + l16;
#pragma unroll
            for (int nt = 0; nt < 4; ++nt)
                o[nt * 16] = acc[mt][nt][r] + bs[nt];
        }
    }
}

extern "C" void kernel_launch(void* const* d_in, const int* in_sizes, int n_in,
                              void* d_out, int out_size, void* d_ws, size_t ws_size,
                              hipStream_t stream) {
    const float* x    = (const float*)d_in[0];
    const float* W    = (const float*)d_in[1];
    const float* bias = (const float*)d_in[2];
    float* out        = (float*)d_out;

    unsigned char* X8  = (unsigned char*)d_ws;
    unsigned char* W8t = X8 + (size_t)BROWS * DDIM;

    {
        const size_t nElem = (size_t)BROWS * DDIM;
        const int threads = 256;
        const int blocks  = (int)(nElem / (threads * 16));
        bin_x_kernel<<<blocks, threads, 0, stream>>>(x, X8);
    }
    {
        dim3 bt(32, 8);
        dim3 gt(DDIM / 32, DDIM / 32);
        bin_wT_kernel<<<gt, bt, 0, stream>>>(W, W8t);
    }
    {
        dim3 gg(BROWS / 128, DDIM / 128);
        bin_gemm_kernel<<<gg, 256, 0, stream>>>(X8, W8t, bias, out);
    }
}
